// DeformableTransformerDecoderLayer_2327872274640
// MI455X (gfx1250) — matrix operand order repaired and checked by reading, not compiled, not run on hardware
//
#include <hip/hip_runtime.h>
#include <hip/hip_bf16.h>

// ---------------------------------------------------------------------------
// MI455X (gfx1250) implementation of a Deformable-DETR decoder layer.
// Dense GEMMs: v_wmma_f32_16x16x32_bf16 (f32 in memory, bf16 in the matrix
// pipe via v_cvt_pk_bf16_f32, f32 accumulate) with double-buffered
// GLOBAL_LOAD_ASYNC_TO_LDS_B128 tile staging synchronized via s_wait_asynccnt.
// Attention: flash-style fused kernel staging the whole per-head K/V
// (2 x 64KB bf16) in the 320KB WGP LDS.
// ---------------------------------------------------------------------------

typedef __attribute__((ext_vector_type(16))) __bf16 v16bf;
typedef __attribute__((ext_vector_type(2)))  __bf16 bf16x2;
typedef __attribute__((ext_vector_type(8)))  float  f32x8;
typedef __attribute__((ext_vector_type(2)))  float  f32x2;

union AFrag { v16bf v; unsigned int u[8]; };

// native f32 -> bf16 (RTNE fptrunc; hardware cvt on gfx1250)
__device__ __forceinline__ unsigned short f2bf(float f) {
  return __builtin_bit_cast(unsigned short, (__bf16)f);
}
// packed pair convert: lowers to a single v_cvt_pk_bf16_f32
__device__ __forceinline__ unsigned int pack2(float lo, float hi) {
  f32x2 f = {lo, hi};
  bf16x2 b = __builtin_convertvector(f, bf16x2);
  return __builtin_bit_cast(unsigned int, b);
}
__device__ __forceinline__ f32x8 wmma_bf16(const AFrag& a, const AFrag& b, f32x8 c) {
  return __builtin_amdgcn_wmma_f32_16x16x32_bf16(false, a.v, false, b.v,
                                                 (short)0, c, false, false);
}
// reductions across the 16 lanes of a half-wave (row is striped over 16 lanes)
__device__ __forceinline__ float rowmax16(float v) {
  for (int m = 1; m < 16; m <<= 1) v = fmaxf(v, __shfl_xor(v, m, 32));
  return v;
}
__device__ __forceinline__ float rowsum16(float v) {
  for (int m = 1; m < 16; m <<= 1) v += __shfl_xor(v, m, 32);
  return v;
}
// LDS byte offset (allocation-relative) of a shared-memory pointer:
// generic LDS addresses carry the offset in bits [31:0] (aperture in [63:32]).
__device__ __forceinline__ unsigned lds_off(const void* p) {
  return (unsigned)(size_t)p;
}

// ---------------------------------------------------------------------------
// Generic GEMM: C[M,N] = act(A[M,K] @ W[N,K]^T + bias) (+mask) (+residual)
// Block = 128 threads (4 waves); tile = 64(M) x 64(N); K step 32.
// M, N multiples of 64; K multiple of 32 (true for every call in this layer).
// All 4 B-fragments are materialized before the 4 back-to-back WMMAs so the
// DS loads batch (staggered dscnt waits) and no WMMA->VALU WAR hazard nops
// are needed.
// ---------------------------------------------------------------------------
__global__ __launch_bounds__(128) void gemm_kernel(
    const float* __restrict__ A, const float* __restrict__ W,
    const float* __restrict__ bias, const float* __restrict__ residual,
    const unsigned char* __restrict__ rowmask,
    float* __restrict__ C, int K, int N, int relu)
{
  __shared__ __align__(16) float As[2][64 * 32];
  __shared__ __align__(16) float Ws[2][64 * 32];

  const int tid  = threadIdx.x;
  const int wave = tid >> 5, lane = tid & 31;
  const int l16  = lane & 15, half = lane >> 4;
  const int row0 = blockIdx.y * 64;
  const int col0 = blockIdx.x * 64;

  auto stage = [&](int buf, int kt) {
    #pragma unroll
    for (int i = 0; i < 4; ++i) {
      int e = (i * 128 + tid) * 4;          // 4-float chunk, covers 2048 f32
      int r = e >> 5, c = e & 31;
      const float* ga = A + (size_t)(row0 + r) * K + kt + c;
      const float* gw = W + (size_t)(col0 + r) * K + kt + c;
      unsigned la = lds_off(&As[buf][e]);
      unsigned lw = lds_off(&Ws[buf][e]);
      asm volatile("global_load_async_to_lds_b128 %0, %1, off"
                   :: "v"(la), "v"(ga) : "memory");
      asm volatile("global_load_async_to_lds_b128 %0, %1, off"
                   :: "v"(lw), "v"(gw) : "memory");
    }
  };

  f32x8 zero = {0.f,0.f,0.f,0.f,0.f,0.f,0.f,0.f};
  f32x8 acc[4];
  #pragma unroll
  for (int i = 0; i < 4; ++i) acc[i] = zero;

  stage(0, 0);
  const int nsteps = K >> 5;
  for (int t = 0; t < nsteps; ++t) {
    const int buf = t & 1;
    if (t + 1 < nsteps) {
      stage(buf ^ 1, (t + 1) * 32);         // prefetch next tile
      asm volatile("s_wait_asynccnt 0x8" ::: "memory");  // current tile done
    } else {
      asm volatile("s_wait_asynccnt 0x0" ::: "memory");
    }
    __syncthreads();

    // A fragment (shared by all 4 N-subtiles of this wave)
    AFrag a;
    const int arow = wave * 16 + l16;
    #pragma unroll
    for (int r = 0; r < 8; ++r) {
      int k = (r < 4 ? 2 * r : 8 + 2 * r) + half * 8;
      a.u[r] = pack2(As[buf][arow * 32 + k], As[buf][arow * 32 + k + 1]);
    }
    // All 4 B fragments in disjoint registers...
    AFrag bfr[4];
    #pragma unroll
    for (int ns = 0; ns < 4; ++ns) {
      const int bcol = ns * 16 + l16;
      #pragma unroll
      for (int r = 0; r < 8; ++r) {
        int k = (r < 4 ? 2 * r : 8 + 2 * r) + half * 8;
        bfr[ns].u[r] = pack2(Ws[buf][bcol * 32 + k], Ws[buf][bcol * 32 + k + 1]);
      }
    }
    // ...then 4 back-to-back WMMAs (no operand overlap -> no hazard nops)
    #pragma unroll
    for (int ns = 0; ns < 4; ++ns)
      acc[ns] = wmma_bf16(a, bfr[ns], acc[ns]);

    __syncthreads();   // all waves done reading buf before it is overwritten
  }

  #pragma unroll
  for (int ns = 0; ns < 4; ++ns) {
    #pragma unroll
    for (int r = 0; r < 8; ++r) {
      int m = row0 + wave * 16 + r + half * 8;
      int n = col0 + ns * 16 + l16;
      float v = acc[ns][r];
      if (bias) v += bias[n];
      if (relu) v = fmaxf(v, 0.f);
      if (rowmask && rowmask[m]) v = 0.f;
      if (residual) v += residual[(size_t)m * N + n];
      C[(size_t)m * N + n] = v;
    }
  }
}

// ---------------------------------------------------------------------------
// LayerNorm over DIM=256; one wave per row; optionally q = h + pos.
// ---------------------------------------------------------------------------
__global__ __launch_bounds__(256) void ln_kernel(
    const float* __restrict__ x, const float* __restrict__ gw,
    const float* __restrict__ gb, const float* __restrict__ pos,
    float* __restrict__ h, float* __restrict__ q, int M)
{
  int row  = blockIdx.x * 8 + (threadIdx.x >> 5);
  int lane = threadIdx.x & 31;
  if (row >= M) return;
  const float* xr = x + (size_t)row * 256;
  float vals[8], s = 0.f, sq = 0.f;
  #pragma unroll
  for (int i = 0; i < 8; ++i) {
    float v = xr[lane + i * 32];
    vals[i] = v; s += v; sq += v * v;
  }
  for (int m = 1; m < 32; m <<= 1) { s += __shfl_xor(s, m, 32); sq += __shfl_xor(sq, m, 32); }
  float mean = s * (1.f / 256.f);
  float var  = sq * (1.f / 256.f) - mean * mean;
  float rs   = rsqrtf(var + 1e-5f);
  #pragma unroll
  for (int i = 0; i < 8; ++i) {
    int c = lane + i * 32;
    float hv = (vals[i] - mean) * rs * gw[c] + gb[c];
    h[(size_t)row * 256 + c] = hv;
    if (q) q[(size_t)row * 256 + c] = hv + pos[(size_t)row * 256 + c];
  }
}

// ---------------------------------------------------------------------------
// Flash attention for one (batch, head, 128-query block).
// qk: [B*LQ, 512] (cols 0..255 = Q heads, 256..511 = K heads); v: [B*LQ, 256].
// LDS: Kt[1024][32] bf16 + Vt[32][1024] bf16 + P scratch[8][16][32] = 136 KB.
// ---------------------------------------------------------------------------
__global__ __launch_bounds__(256) void attn_kernel(
    const float* __restrict__ qk, const float* __restrict__ v,
    float* __restrict__ o)
{
  extern __shared__ unsigned short smem[];
  unsigned short* Kt = smem;                  // [1024][32]
  unsigned short* Vt = smem + 1024 * 32;      // [32][1024] (transposed)
  unsigned short* Ps = smem + 2 * 1024 * 32;  // [8 waves][16][32]

  const int tid  = threadIdx.x;
  const int wave = tid >> 5, lane = tid & 31;
  const int l16  = lane & 15, half = lane >> 4;

  int blk = blockIdx.x;
  const int qb = blk & 7;  blk >>= 3;
  const int hh = blk & 7;  blk >>= 3;
  const int b  = blk;
  const size_t base = (size_t)b * 1024;
  const float kscale = 0.17677669529663687f;   // 1/sqrt(HD)

  for (int i = 0; i < 128; ++i) {              // 32768 elems / 256 thr
    int e = i * 256 + tid;
    int key = e >> 5, d = e & 31;
    Kt[key * 32 + d]   = f2bf(qk[(base + key) * 512 + 256 + hh * 32 + d] * kscale);
    Vt[d * 1024 + key] = f2bf(v[(base + key) * 256 + hh * 32 + d]);
  }
  __syncthreads();

  const int qrow0 = qb * 128 + wave * 16;
  AFrag qa;
  {
    const float* qp = qk + (base + qrow0 + l16) * 512 + hh * 32;
    #pragma unroll
    for (int r = 0; r < 8; ++r) {
      int k = (r < 4 ? 2 * r : 8 + 2 * r) + half * 8;
      qa.u[r] = pack2(qp[k], qp[k + 1]);
    }
  }

  float mprev[8], lsum[8];
  f32x8 zero = {0.f,0.f,0.f,0.f,0.f,0.f,0.f,0.f};
  f32x8 o0 = zero, o1 = zero;
  #pragma unroll
  for (int r = 0; r < 8; ++r) { mprev[r] = -1e30f; lsum[r] = 0.f; }
  unsigned short* Pw = Ps + wave * 16 * 32;

  for (int jb = 0; jb < 32; ++jb) {            // 32 keys per step
    const int key0 = jb * 32;
    AFrag kb0, kb1;
    #pragma unroll
    for (int r = 0; r < 8; ++r) {
      int k = (r < 4 ? 2 * r : 8 + 2 * r) + half * 8;
      kb0.u[r] = *(const unsigned int*)&Kt[(key0 + l16) * 32 + k];
      kb1.u[r] = *(const unsigned int*)&Kt[(key0 + 16 + l16) * 32 + k];
    }
    f32x8 s0 = wmma_bf16(qa, kb0, zero);
    f32x8 s1 = wmma_bf16(qa, kb1, zero);

    #pragma unroll
    for (int r = 0; r < 8; ++r) {
      float rm = rowmax16(fmaxf(s0[r], s1[r]));
      float nm = fmaxf(mprev[r], rm);
      float alpha = __expf(mprev[r] - nm);
      mprev[r] = nm;
      float p0 = __expf(s0[r] - nm);
      float p1 = __expf(s1[r] - nm);
      lsum[r] = lsum[r] * alpha + rowsum16(p0 + p1);
      o0[r] *= alpha; o1[r] *= alpha;
      int prow = r + half * 8;
      Pw[prow * 32 + l16]      = f2bf(p0);
      Pw[prow * 32 + 16 + l16] = f2bf(p1);
    }
    AFrag pa;
    #pragma unroll
    for (int r = 0; r < 8; ++r) {
      int k = (r < 4 ? 2 * r : 8 + 2 * r) + half * 8;
      pa.u[r] = *(const unsigned int*)&Pw[l16 * 32 + k];
    }
    AFrag vb0, vb1;
    #pragma unroll
    for (int r = 0; r < 8; ++r) {
      int k = key0 + (r < 4 ? 2 * r : 8 + 2 * r) + half * 8;
      vb0.u[r] = *(const unsigned int*)&Vt[l16 * 1024 + k];
      vb1.u[r] = *(const unsigned int*)&Vt[(16 + l16) * 1024 + k];
    }
    o0 = wmma_bf16(pa, vb0, o0);
    o1 = wmma_bf16(pa, vb1, o1);
  }

  #pragma unroll
  for (int r = 0; r < 8; ++r) {
    int row = qrow0 + r + half * 8;
    float inv = 1.f / lsum[r];
    o[(base + row) * 256 + hh * 32 + l16]      = o0[r] * inv;
    o[(base + row) * 256 + hh * 32 + 16 + l16] = o1[r] * inv;
  }
}

// ---------------------------------------------------------------------------
// Deformable sampling: one wave per (b, q, head); lane = HD channel.
// Also writes the `loc` output tensor [B,LQ,NH,NL,NP,2].
// ---------------------------------------------------------------------------
__global__ __launch_bounds__(256) void deform_kernel(
    const float* __restrict__ offb, const float* __restrict__ awb,
    const float* __restrict__ valueb, const float* __restrict__ ref,
    float* __restrict__ samp, float* __restrict__ loc_out)
{
  int wid  = blockIdx.x * 8 + (threadIdx.x >> 5);
  int lane = threadIdx.x & 31;
  const int hh = wid & 7;  wid >>= 3;
  const int q  = wid & 1023; wid >>= 10;
  const int b  = wid;
  const size_t row = (size_t)b * 1024 + q;
  const int ST[4] = {0, 4096, 5120, 5376};

  float aw[16];
  float mx = -1e30f;
  #pragma unroll
  for (int j = 0; j < 16; ++j) { aw[j] = awb[row * 128 + hh * 16 + j]; mx = fmaxf(mx, aw[j]); }
  float sum = 0.f;
  #pragma unroll
  for (int j = 0; j < 16; ++j) { aw[j] = __expf(aw[j] - mx); sum += aw[j]; }
  const float inv = 1.f / sum;

  const float* vb = valueb + (size_t)b * 5440 * 256 + hh * 32 + lane;
  float acc = 0.f;
  #pragma unroll
  for (int l = 0; l < 4; ++l) {
    const int Hl = 64 >> l, Wl = 64 >> l;
    const int st = ST[l];
    float rx = ref[(row * 4 + l) * 2 + 0];
    float ry = ref[(row * 4 + l) * 2 + 1];
    #pragma unroll
    for (int p = 0; p < 4; ++p) {
      float ox = offb[row * 256 + hh * 32 + l * 8 + p * 2 + 0];
      float oy = offb[row * 256 + hh * 32 + l * 8 + p * 2 + 1];
      float lx = rx + ox / (float)Wl;
      float ly = ry + oy / (float)Hl;
      if (lane == 0) {
        size_t li = (((row * 8 + hh) * 4 + l) * 4 + p) * 2;
        loc_out[li] = lx; loc_out[li + 1] = ly;
      }
      float px = lx * Wl - 0.5f, py = ly * Hl - 0.5f;
      float x0f = floorf(px), y0f = floorf(py);
      float fx = px - x0f, fy = py - y0f;
      int x0 = (int)x0f, y0 = (int)y0f;
      float sv = 0.f;
      #pragma unroll
      for (int cy = 0; cy < 2; ++cy)
        #pragma unroll
        for (int cx = 0; cx < 2; ++cx) {
          int xi = x0 + cx, yi = y0 + cy;
          if (xi >= 0 && xi < Wl && yi >= 0 && yi < Hl) {
            float wgt = (cx ? fx : 1.f - fx) * (cy ? fy : 1.f - fy);
            sv += wgt * vb[(size_t)(st + yi * Wl + xi) * 256];
          }
        }
      acc += sv * aw[l * 4 + p] * inv;
    }
  }
  samp[row * 256 + hh * 32 + lane] = acc;
}

// ---------------------------------------------------------------------------
extern "C" void kernel_launch(void* const* d_in, const int* in_sizes, int n_in,
                              void* d_out, int out_size, void* d_ws, size_t ws_size,
                              hipStream_t stream) {
  const float* input  = (const float*)d_in[0];
  const float* pos    = (const float*)d_in[1];
  const float* refpt  = (const float*)d_in[2];
  const float* source = (const float*)d_in[3];
  const unsigned char* mask = (const unsigned char*)d_in[6];
  const float* snw = (const float*)d_in[7],  *snb = (const float*)d_in[8];
  const float* cnw = (const float*)d_in[9],  *cnb = (const float*)d_in[10];
  const float* fnw = (const float*)d_in[11], *fnb = (const float*)d_in[12];
  const float* w_qkv = (const float*)d_in[13], *b_qkv = (const float*)d_in[14];
  const float* w_mo  = (const float*)d_in[15], *b_mo  = (const float*)d_in[16];
  const float* w_off = (const float*)d_in[17], *b_off = (const float*)d_in[18];
  const float* w_aw  = (const float*)d_in[19], *b_aw  = (const float*)d_in[20];
  const float* w_val = (const float*)d_in[21], *b_val = (const float*)d_in[22];
  const float* w_out = (const float*)d_in[23], *b_out = (const float*)d_in[24];
  const float* w_f1  = (const float*)d_in[25], *b_f1  = (const float*)d_in[26];
  const float* w_f2  = (const float*)d_in[27], *b_f2  = (const float*)d_in[28];

  float* out_x   = (float*)d_out;                  // [16384, 256]
  float* out_loc = out_x + (size_t)16384 * 256;    // [B,LQ,NH,NL,NP,2]

  char* ws = (char*)d_ws;
  size_t off = 0;
  auto alloc = [&](size_t bytes) {
    void* p = ws + off;
    off += (bytes + 255) & ~(size_t)255;
    return p;
  };
  const int M = 16384;                             // B*LQ
  float* hbuf  = (float*)alloc((size_t)M * 256 * 4);
  float* qbuf  = (float*)alloc((size_t)M * 256 * 4);
  float* ffbuf = (float*)alloc((size_t)M * 1024 * 4);  // qkbuf aliases here
  float* qkbuf = ffbuf;                                // dead before ff1 runs
  float* vbuf  = (float*)alloc((size_t)M * 256 * 4);   // offbuf aliases (vbuf dead after attn)
  float* offbuf = vbuf;
  float* attnbuf = (float*)alloc((size_t)M * 256 * 4); // sampbuf aliases
  float* sampbuf = attnbuf;
  float* xbuf  = (float*)alloc((size_t)M * 256 * 4);
  float* awbuf = (float*)alloc((size_t)M * 128 * 4);
  float* valbuf = (float*)alloc((size_t)87040 * 256 * 4);

  // 1) h = LN(input); q = h + pos
  ln_kernel<<<M / 8, 256, 0, stream>>>(input, snw, snb, pos, hbuf, qbuf, M);
  // 2) Q,K projection (rows 0..511 of mha_in) from q; V projection (rows 512..767) from h
  gemm_kernel<<<dim3(8, M / 64), 128, 0, stream>>>(qbuf, w_qkv, b_qkv, nullptr, nullptr, qkbuf, 256, 512, 0);
  gemm_kernel<<<dim3(4, M / 64), 128, 0, stream>>>(hbuf, w_qkv + 512 * 256, b_qkv + 512, nullptr, nullptr, vbuf, 256, 256, 0);
  // 3) fused flash attention
  attn_kernel<<<16 * 8 * 8, 256, 139264, stream>>>(qkbuf, vbuf, attnbuf);
  // 4) x = input + attn @ Wout^T + b
  gemm_kernel<<<dim3(4, M / 64), 128, 0, stream>>>(attnbuf, w_mo, b_mo, input, nullptr, xbuf, 256, 256, 0);
  // 5) h2 = LN(x); q2 = h2 + pos
  ln_kernel<<<M / 8, 256, 0, stream>>>(xbuf, cnw, cnb, pos, hbuf, qbuf, M);
  // 6) offsets / attention-weight / value projections
  gemm_kernel<<<dim3(4, M / 64), 128, 0, stream>>>(qbuf, w_off, b_off, nullptr, nullptr, offbuf, 256, 256, 0);
  gemm_kernel<<<dim3(2, M / 64), 128, 0, stream>>>(qbuf, w_aw, b_aw, nullptr, nullptr, awbuf, 256, 128, 0);
  gemm_kernel<<<dim3(4, 87040 / 64), 128, 0, stream>>>(source, w_val, b_val, nullptr, mask, valbuf, 256, 256, 0);
  // 7) deformable bilinear sampling (+ loc output)
  deform_kernel<<<(16 * 1024 * 8) / 8, 256, 0, stream>>>(offbuf, awbuf, valbuf, refpt, sampbuf, out_loc);
  // 8) x = x + samp @ Wout2^T + b   (in-place residual is element-local -> safe)
  gemm_kernel<<<dim3(4, M / 64), 128, 0, stream>>>(sampbuf, w_out, b_out, xbuf, nullptr, xbuf, 256, 256, 0);
  // 9) h3 = LN(x)
  ln_kernel<<<M / 8, 256, 0, stream>>>(xbuf, fnw, fnb, nullptr, hbuf, nullptr, M);
  // 10) FFN
  gemm_kernel<<<dim3(16, M / 64), 128, 0, stream>>>(hbuf, w_f1, b_f1, nullptr, nullptr, ffbuf, 256, 1024, 1);
  gemm_kernel<<<dim3(4, M / 64), 128, 0, stream>>>(ffbuf, w_f2, b_f2, xbuf, nullptr, out_x, 1024, 256, 0);
}
